// PSWarpHead_18167711662481
// MI455X (gfx1250) — matrix operand order repaired and checked by reading, not compiled, not run on hardware
//
#include <hip/hip_runtime.h>
#include <math.h>

// ---------------------------------------------------------------------------
// PSWarpHead on MI455X (gfx1250): fused 3x3 conv (implicit GEMM via
// v_wmma_f32_16x16x32_bf16) + BN + ReLU + 1x1 conv, then rotated-window
// bilinear sampling.  Each wave owns TWO 16x16 accumulators sharing one
// B fragment -> halves L2 weight traffic, 2 independent WMMA chains/wave.
// ---------------------------------------------------------------------------

#define B_    4
#define CIN   256
#define H_    200
#define W_    176
#define COUT  28
#define COUTP 32          // padded to 32 for two 16-wide N tiles
#define NBOX  2048
#define WIN_H 4
#define WIN_W 7
#define WIN_  (WIN_H * WIN_W)
#define BN_EPS 1e-3f

#define TILE_PIX 64       // pixels (along W) per workgroup
#define WTILES   3        // ceil(176/64)
#define KCHUNKS  8        // 256 / 32 input-channel chunks
#define CPAD     40       // inner LDS padding (80B col stride: 16B-aligned b128,
                          // 20-bank stride -> conflict-free across 16 lanes)
#define BLOCK_T  128      // 4 waves = 2 M-pairs x 2 N-tiles

typedef __bf16 bf16;
typedef __attribute__((ext_vector_type(16))) __bf16 v16bf;
typedef __attribute__((ext_vector_type(8)))  float  v8f;

union Frag16 { v16bf v; uint4 q[2]; };

// ---------------------------------------------------------------------------
// Kernel 0: repack w1 (BN scale folded) -> [tap][chunk][co(32)][k(32)] bf16,
//           and w2 -> [co(32)][k(32)] bf16 (zero padded).
// ---------------------------------------------------------------------------
__global__ __launch_bounds__(256) void pack_weights_kernel(
    const float* __restrict__ w1, const float* __restrict__ gamma,
    const float* __restrict__ var, const float* __restrict__ w2,
    bf16* __restrict__ w1p, bf16* __restrict__ w2p) {
  const int i = blockIdx.x * 256 + threadIdx.x;
  const int N1 = 9 * KCHUNKS * COUTP * 32;      // 73728
  if (i < N1) {
    const int k   = i & 31;
    const int co  = (i >> 5) & 31;
    const int cc  = (i >> 10) & 7;
    const int tap = i >> 13;
    float val = 0.f;
    if (co < COUT) {
      const int ci = cc * 32 + k;
      const int kh = tap / 3, kw = tap % 3;
      const float sc = gamma[co] * rsqrtf(var[co] + BN_EPS);
      val = w1[((co * CIN + ci) * 3 + kh) * 3 + kw] * sc;
    }
    w1p[i] = (bf16)val;
  } else if (i < N1 + COUTP * 32) {
    const int j  = i - N1;
    const int k  = j & 31;
    const int co = j >> 5;
    float val = (co < COUT && k < COUT) ? w2[co * COUT + k] : 0.f;
    w2p[j] = (bf16)val;
  }
}

// ---------------------------------------------------------------------------
// Kernel 1: fused conv3x3 + bias + ReLU + conv1x1 -> feat[B][28][H][W]
// One block = 64 pixels of one (b, h) row x all 28 output channels.
// 4 waves; each wave = two 16-pixel M-subtiles x one 16-channel N-tile.
// Main loop: 8 chunks x 9 taps x (2 A ds_b128-pairs + 1 B global pair +
// 2 WMMAs). Epilogue: bias+ReLU -> LDS bf16 -> 2 WMMAs for the 1x1 conv.
// ---------------------------------------------------------------------------
__global__ __launch_bounds__(BLOCK_T) void conv_fused_kernel(
    const float* __restrict__ x, const bf16* __restrict__ w1p,
    const bf16* __restrict__ w2p, const float* __restrict__ gamma,
    const float* __restrict__ beta, const float* __restrict__ mean,
    const float* __restrict__ var, float* __restrict__ feat) {

  __shared__ __align__(16) bf16 xsh[3][TILE_PIX + 2][CPAD];  // [row][col][ci]
  __shared__ __align__(16) bf16 ylds[TILE_PIX][CPAD];        // [pix][ych]

  const int wg = blockIdx.x;
  const int wt = wg % WTILES;
  const int h  = (wg / WTILES) % H_;
  const int b  = wg / (WTILES * H_);
  const int w0 = wt * TILE_PIX;

  const int tid  = threadIdx.x;
  const int lane = tid & 31;
  const int wave = tid >> 5;         // 0..3
  const int mp   = wave >> 1;        // 0..1 : 32-pixel M pair
  const int nt   = wave & 1;         // 0..1 : channel tile
  const int hi   = lane >> 4;        // lane half
  const int l15  = lane & 15;
  const int mBase0 = mp * 32 + l15;  // A-row pixel for subtile 0
  const int co     = nt * 16 + l15;  // output channel for B / C columns
  const int hiA    = 8 * hi;

  v8f acc0 = (v8f)0.0f;
  v8f acc1 = (v8f)0.0f;

  for (int cc = 0; cc < KCHUNKS; ++cc) {
    __syncthreads();
    // ---- stage x chunk: 32 ci x 3 rows x 66 cols, f32 -> bf16, zero-pad ----
    for (int e = tid; e < 3 * 66 * 32; e += BLOCK_T) {
      const int col = e % 66;
      const int t2  = e / 66;
      const int ch  = t2 % 32;
      const int row = t2 / 32;
      const int gw = w0 + col - 1;
      const int gh = h + row - 1;
      float v = 0.f;
      if (gw >= 0 && gw < W_ && gh >= 0 && gh < H_) {
        v = x[((b * CIN + cc * 32 + ch) * H_ + gh) * W_ + gw];
      }
      xsh[row][col][ch] = (bf16)v;
    }
    __syncthreads();

    // ---- 9 taps x two 16x16x32 WMMAs (shared B fragment) ----
#pragma unroll
    for (int tap = 0; tap < 9; ++tap) {
      const int kh = tap / 3, kw = tap % 3;
      Frag16 a0, a1, bfg;
      // A: per-lane two contiguous 8xbf16 runs (ISA A-layout: K=(e&7)+16*(e>>3)+8*hi)
      const bf16* ap0 = &xsh[kh][mBase0 + kw][0];
      const bf16* ap1 = &xsh[kh][mBase0 + 16 + kw][0];
      a0.q[0] = *(const uint4*)(ap0 + hiA);
      a0.q[1] = *(const uint4*)(ap0 + 16 + hiA);
      a1.q[0] = *(const uint4*)(ap1 + hiA);
      a1.q[1] = *(const uint4*)(ap1 + 16 + hiA);
      // B: per-lane two 8xbf16 runs (same K order as A: K=(e&7)+16*(e>>3)+8*hi), L2-resident
      const bf16* wp = w1p + (((tap * KCHUNKS + cc) * COUTP + co) * 32 + 8 * hi);
      bfg.q[0] = ((const uint4*)wp)[0];
      bfg.q[1] = ((const uint4*)wp)[2];
      acc0 = __builtin_amdgcn_wmma_f32_16x16x32_bf16(
          false, a0.v, false, bfg.v, (short)0, acc0, false, false);
      acc1 = __builtin_amdgcn_wmma_f32_16x16x32_bf16(
          false, a1.v, false, bfg.v, (short)0, acc1, false, false);
    }
  }

  // ---- BN bias + ReLU, spill y tile (bf16) to LDS for the 1x1 conv ----
  float biasv = 0.f;
  const bool nvalid = (co < COUT);
  if (nvalid) {
    const float sc = gamma[co] * rsqrtf(var[co] + BN_EPS);
    biasv = beta[co] - mean[co] * sc;
  }
  __syncthreads();
#pragma unroll
  for (int v = 0; v < 8; ++v) {
    const int m = v + 8 * hi;                 // C/D layout: M = v + 8*laneHi
    const float y0 = nvalid ? fmaxf(acc0[v] + biasv, 0.f) : 0.f;
    const float y1 = nvalid ? fmaxf(acc1[v] + biasv, 0.f) : 0.f;
    ylds[mp * 32 + m][co]      = (bf16)y0;
    ylds[mp * 32 + 16 + m][co] = (bf16)y1;
  }
  __syncthreads();

  // ---- 1x1 conv: out = y(64x32) x w2(32x32), two WMMAs per wave ----
  Frag16 a0, a1, bf2;
  const bf16* yp0 = &ylds[mBase0][0];
  const bf16* yp1 = &ylds[mBase0 + 16][0];
  a0.q[0] = *(const uint4*)(yp0 + hiA);
  a0.q[1] = *(const uint4*)(yp0 + 16 + hiA);
  a1.q[0] = *(const uint4*)(yp1 + hiA);
  a1.q[1] = *(const uint4*)(yp1 + 16 + hiA);
  const bf16* w2l = w2p + co * 32 + 8 * hi;
  bf2.q[0] = ((const uint4*)w2l)[0];
  bf2.q[1] = ((const uint4*)w2l)[2];
  v8f out0 = (v8f)0.0f, out1 = (v8f)0.0f;
  out0 = __builtin_amdgcn_wmma_f32_16x16x32_bf16(
      false, a0.v, false, bf2.v, (short)0, out0, false, false);
  out1 = __builtin_amdgcn_wmma_f32_16x16x32_bf16(
      false, a1.v, false, bf2.v, (short)0, out1, false, false);

  // ---- store feat[b][co][h][w] ----
  if (co < COUT) {
    const float* base = feat;  // silence unused warnings pattern
    (void)base;
#pragma unroll
    for (int v = 0; v < 8; ++v) {
      const int m = v + 8 * hi;
      const int gw0 = w0 + mp * 32 + m;
      const int gw1 = gw0 + 16;
      const int rowb = ((b * COUT + co) * H_ + h) * W_;
      if (gw0 < W_) feat[rowb + gw0] = out0[v];
      if (gw1 < W_) feat[rowb + gw1] = out1[v];
    }
  }
}

// ---------------------------------------------------------------------------
// Kernel 2: rotated sample grid + bilinear PS-warp sampling + mean over the
// 4x7 window. One thread per (b, n) box.
// ---------------------------------------------------------------------------
__global__ __launch_bounds__(256) void pswarp_sample_kernel(
    const float* __restrict__ feat, const float* __restrict__ ga,
    float* __restrict__ out) {
  const int idx = blockIdx.x * 256 + threadIdx.x;
  if (idx >= B_ * NBOX) return;
  const int b = idx / NBOX;

  const float* a = ga + idx * 7;
  const float xg = a[0], yg = a[1], wg = a[3], lg = a[4], rg = a[6];
  const float c = cosf(rg), s = sinf(rg);

  float sum = 0.f;
#pragma unroll
  for (int i = 0; i < WIN_H; ++i) {
    const float xx = ((float)i / (WIN_H - 1) - 0.5f) * wg;
#pragma unroll
    for (int j = 0; j < WIN_W; ++j) {
      const float yy = ((float)j / (WIN_W - 1) - 0.5f) * lg;
      const float xpix = xx * c + yy * s + xg;
      const float ypix = yy * c - xx * s + yg;
      const float xs = (xpix + 0.0f)  * 2.5f;   // GRID_OFF.x, 1/0.4
      const float ys = (ypix + 40.0f) * 2.5f;   // GRID_OFF.y

      const float x0 = floorf(xs), y0 = floorf(ys);
      const float wx = xs - x0, wy = ys - y0;
      const int ch = i * WIN_W + j;
      const float* img = feat + (b * COUT + ch) * (H_ * W_);

      float v00 = 0.f, v01 = 0.f, v10 = 0.f, v11 = 0.f;
      const float x1 = x0 + 1.f, y1 = y0 + 1.f;
      const int xi0 = min(max((int)x0, 0), W_ - 1);
      const int xi1 = min(max((int)x1, 0), W_ - 1);
      const int yi0 = min(max((int)y0, 0), H_ - 1);
      const int yi1 = min(max((int)y1, 0), H_ - 1);
      const bool vx0 = (x0 >= 0.f) && (x0 <= (float)(W_ - 1));
      const bool vx1 = (x1 >= 0.f) && (x1 <= (float)(W_ - 1));
      const bool vy0 = (y0 >= 0.f) && (y0 <= (float)(H_ - 1));
      const bool vy1 = (y1 >= 0.f) && (y1 <= (float)(H_ - 1));
      if (vy0 && vx0) v00 = img[yi0 * W_ + xi0];
      if (vy0 && vx1) v01 = img[yi0 * W_ + xi1];
      if (vy1 && vx0) v10 = img[yi1 * W_ + xi0];
      if (vy1 && vx1) v11 = img[yi1 * W_ + xi1];

      sum += v00 * (1.f - wx) * (1.f - wy) + v01 * wx * (1.f - wy) +
             v10 * (1.f - wx) * wy + v11 * wx * wy;
    }
  }
  out[idx] = sum * (1.0f / (float)WIN_);
}

// ---------------------------------------------------------------------------
extern "C" void kernel_launch(void* const* d_in, const int* in_sizes, int n_in,
                              void* d_out, int out_size, void* d_ws, size_t ws_size,
                              hipStream_t stream) {
  const float* x     = (const float*)d_in[0];
  const float* ga    = (const float*)d_in[1];
  const float* w1    = (const float*)d_in[2];
  const float* gamma = (const float*)d_in[3];
  const float* beta  = (const float*)d_in[4];
  const float* mean  = (const float*)d_in[5];
  const float* var   = (const float*)d_in[6];
  const float* w2    = (const float*)d_in[7];
  float* out = (float*)d_out;

  char* ws = (char*)d_ws;
  bf16* w1p  = (bf16*)(ws);                         // 9*8*32*32*2 = 147456 B
  bf16* w2p  = (bf16*)(ws + 147456);                // 32*32*2     =   2048 B
  float* feat = (float*)(ws + 149504);              // 4*28*200*176*4 = 15.77 MB

  // 0) repack/fold weights to bf16 WMMA layouts
  const int packN = 9 * KCHUNKS * COUTP * 32 + COUTP * 32;
  pack_weights_kernel<<<(packN + 255) / 256, 256, 0, stream>>>(
      w1, gamma, var, w2, w1p, w2p);

  // 1) fused conv3x3 + BN + ReLU + conv1x1 (WMMA implicit GEMM)
  conv_fused_kernel<<<B_ * H_ * WTILES, BLOCK_T, 0, stream>>>(
      x, w1p, w2p, gamma, beta, mean, var, feat);

  // 2) rotated-window bilinear sampling + mean
  pswarp_sample_kernel<<<(B_ * NBOX + 255) / 256, 256, 0, stream>>>(
      feat, ga, out);
}
